// CoPE2d_68925635167047
// MI455X (gfx1250) — matrix operand order repaired and checked by reading, not compiled, not run on hardware
//
#include <hip/hip_runtime.h>

#define NSEQ   256
#define CDIM   64
#define NPOSM  288
#define NTILES 18
#define QSTR   68      // padded qtile stride (floats)
#define SSTR   260     // padded sig/pos_h stride (floats)

typedef __attribute__((ext_vector_type(16))) _Float16 v16h;
typedef __attribute__((ext_vector_type(8)))  float    v8f;
typedef __attribute__((ext_vector_type(4)))  float    f32x4;

// ---------------------------------------------------------------------------
// Kernel 1: pack pos_emb (C=64 x NPOS=288, row-major) into WMMA B-fragment
// layout (32x16 f16 B tiles), split into f16 hi + lo for near-fp32 accuracy.
// ws layout: hi[18432] halves, then lo[18432] halves (73,728 bytes total).
// Fragment element e of lane `lane`, k-block kb, n-tile t holds
//   pos_emb[ kb*32 + (lane>>4)*16 + e ][ t*16 + (lane&15) ]
// ---------------------------------------------------------------------------
__global__ void cope_pack_pemb(const float* __restrict__ pemb,
                               _Float16* __restrict__ wsb) {
  int u    = blockIdx.x * blockDim.x + threadIdx.x;   // 0..18431
  int e    = u & 15;
  int lane = (u >> 4) & 31;
  int kb   = (u >> 9) & 1;
  int t    = u >> 10;
  int k = kb * 32 + ((lane >> 4) << 4) + e;           // 0..63
  int n = (t << 4) + (lane & 15);                     // 0..287
  float v = pemb[k * NPOSM + n];
  _Float16 hi = (_Float16)v;
  _Float16 lo = (_Float16)(v - (float)hi);
  wsb[u]         = hi;
  wsb[18432 + u] = lo;
}

// ---------------------------------------------------------------------------
// Main kernel: one block per (b*NH + h, 16-row block). 256 threads = 8 waves.
// ---------------------------------------------------------------------------
__global__ void __launch_bounds__(256)
cope_main(const float* __restrict__ query,
          const float* __restrict__ attn,
          const _Float16* __restrict__ wsb,
          float* __restrict__ out) {
  __shared__ float qtile[16 * QSTR];        //  4,352 B
  __shared__ float llog [16 * NPOSM];       // 18,432 B  logits_int tile
  __shared__ float sigt [16 * SSTR];        // 16,640 B  sigmoid(strip)
  __shared__ float phst [16 * SSTR];        // 16,640 B  pos_h(strip)

  const int tid  = threadIdx.x;
  const int wg   = blockIdx.x;              // bh*16 + wblk
  const int wblk = wg & 15;
  const int bhi  = wg >> 4;
  const long qbase = ((long)bhi * NSEQ + wblk * 16) * CDIM;
  const long gbase = ((long)bhi * NSEQ + wblk * 16) * NSEQ;

  // ---- stage 16x64 query tile into LDS (padded stride) -------------------
  {
    f32x4 v = ((const f32x4*)(query + qbase))[tid];
    int m = tid >> 4, off = (tid & 15) << 2;
    *(f32x4*)&qtile[m * QSTR + off] = v;
  }
  __syncthreads();

  // ---- GEMM: llog[16][288] = qtile[16][64] @ pos_emb[64][288] ------------
  const int lane = tid & 31;
  const int wv   = tid >> 5;                // wave id 0..7
  const int mrow = lane & 15;
  const int hlf  = lane >> 4;

  // A fragments (16x32 f16, hi/lo split), identical across tiles
  v16h ah[2], al[2];
#pragma unroll
  for (int kb = 0; kb < 2; ++kb) {
    const float* qp = &qtile[mrow * QSTR + kb * 32 + hlf * 16];
#pragma unroll
    for (int i = 0; i < 8; ++i) {
      float x = qp[i];
      _Float16 h = (_Float16)x;
      ah[kb][i] = h;
      al[kb][i] = (_Float16)(x - (float)h);
      float y = qp[8 + i];
      h = (_Float16)y;
      ah[kb][8 + i] = h;
      al[kb][8 + i] = (_Float16)(y - (float)h);
    }
  }

  for (int t = wv; t < NTILES; t += 8) {
    v8f acc = {};
#pragma unroll
    for (int kb = 0; kb < 2; ++kb) {
      long fo = ((long)(t * 2 + kb) * 32 + lane) << 4;
      v16h bh16 = *(const v16h*)(wsb + fo);
      v16h bl16 = *(const v16h*)(wsb + 18432 + fo);
      acc = __builtin_amdgcn_wmma_f32_16x16x32_f16(false, ah[kb], false, bh16,
                                                   (short)0, acc, false, false);
      acc = __builtin_amdgcn_wmma_f32_16x16x32_f16(false, ah[kb], false, bl16,
                                                   (short)0, acc, false, false);
      acc = __builtin_amdgcn_wmma_f32_16x16x32_f16(false, al[kb], false, bh16,
                                                   (short)0, acc, false, false);
    }
    // D layout: VGPR v -> row v (lanes 0-15) / row v+8 (lanes 16-31), col = lane&15
#pragma unroll
    for (int v = 0; v < 8; ++v) {
      int row = v + hlf * 8;
      llog[row * NPOSM + t * 16 + mrow] = acc[v];
    }
  }
  __syncthreads();

  // ---- Phase A: thread = column j. sigmoid + pos_h (column suffix sum) ---
  {
    float s[16];
    const float* gp = attn + gbase + tid;
#pragma unroll
    for (int i = 0; i < 16; ++i) s[i] = gp[i * NSEQ];      // coalesced rows
#pragma unroll
    for (int i = 0; i < 16; ++i) s[i] = 1.0f / (1.0f + __expf(-s[i]));
    float run = 0.0f;
#pragma unroll
    for (int i = 15; i >= 0; --i) {
      run += s[i];
      sigt[i * SSTR + tid] = s[i];
      phst[i * SSTR + tid] = run;                          // pos_h[i][tid]
    }
  }
  __syncthreads();

  // ---- Phase B: thread = (col-block r, row). pos_w + gather + lerp -------
  {
    const int r = tid >> 4, row = tid & 15;
    float sg[16], ph[16];
#pragma unroll
    for (int q = 0; q < 4; ++q) {
      *(f32x4*)&sg[q * 4] = *(const f32x4*)&sigt[row * SSTR + r * 16 + q * 4];
      *(f32x4*)&ph[q * 4] = *(const f32x4*)&phst[row * SSTR + r * 16 + q * 4];
    }
    float res[16];
    float pw = 0.0f;
#pragma unroll
    for (int c = 15; c >= 0; --c) {
      pw += sg[c];                                         // pos_w suffix sum
      float pos = fminf(ph[c] * 16.0f + pw, 287.0f);
      int   fi  = (int)pos;                                // pos >= 0 -> floor
      float w   = pos - (float)fi;
      int   ci  = fi + 1; if (ci > 287) ci = 287;
      float Lf = llog[row * NPOSM + fi];
      float Lc = llog[row * NPOSM + ci];
      res[c] = Lc * w + Lf * (1.0f - w);
    }
    float* op = out + gbase + row * NSEQ + r * 16;
#pragma unroll
    for (int q = 0; q < 4; ++q)
      ((f32x4*)op)[q] = *(f32x4*)&res[q * 4];
  }
}

extern "C" void kernel_launch(void* const* d_in, const int* in_sizes, int n_in,
                              void* d_out, int out_size, void* d_ws, size_t ws_size,
                              hipStream_t stream) {
  const float* query = (const float*)d_in[0];   // (64,16,256,64)  f32
  const float* attn  = (const float*)d_in[1];   // (64,16,256,256) f32
  const float* pemb  = (const float*)d_in[2];   // (64,288)        f32
  _Float16* wsb = (_Float16*)d_ws;              // 73,728 B used
  float* outp = (float*)d_out;                  // (64,16,256,256) f32

  cope_pack_pemb<<<72, 256, 0, stream>>>(pemb, wsb);
  cope_main<<<64 * 16 * 16, 256, 0, stream>>>(query, attn, wsb, outp);
}
